// RMTLayer_87162066305434
// MI455X (gfx1250) — matrix operand order repaired and checked by reading, not compiled, not run on hardware
//
#include <hip/hip_runtime.h>

// ---------------------------------------------------------------------------
// RMT layer for MI455X (gfx1250, wave32, WMMA).
// One persistent workgroup per batch (16 blocks x 512 threads = 16 waves),
// looping the 64-step recurrence internally. All GEMMs use
// v_wmma_f32_16x16x32_bf16. Weights are repacked once into the WMMA
// B-operand lane layout so every B fragment is one 32-byte contiguous load.
// Projection GEMMs are register-blocked 4x in N: one A fragment feeds four
// WMMAs (loads/WMMA = 1.5), with the 4 B loads issued back-to-back for
// clause formation. Next step's hidden block is prefetched during phase 5.
// ---------------------------------------------------------------------------

typedef __bf16 bf16_t;
typedef __bf16 v16bf __attribute__((ext_vector_type(16)));
typedef __bf16 v8bf  __attribute__((ext_vector_type(8)));
typedef float  v8f   __attribute__((ext_vector_type(8)));

#define B_      16
#define NM_     64
#define MB_     64
#define D_      1024
#define NH_     16
#define NKV_    8
#define MEM_    64
#define HD_     64
#define S_      128
#define KVD_    512      // NKV*HD
#define NWAVE   16
#define NTHREADS 512

// workspace layout (bytes); total required = 6324224 + 16*1310720 = 27,295,744 B (~26 MiB)
#define OFF_WQ   0u         // bf16 packed [32 kt][64 nt][32 lane][16 e]  (2 MiB)
#define OFF_WK   2097152u   // (1 MiB)
#define OFF_WV   3145728u   // (1 MiB)
#define OFF_WO   4194304u   // (2 MiB)
#define OFF_RC   6291456u   // rope cos [128][32] f32 (16 KiB)
#define OFF_RS   6307840u   // rope sin            (16 KiB)
#define OFF_BATCH 6324224u
#define PER_BATCH 1310720u
#define BO_MEM 0u           // f32  [64][1024]   256 KiB
#define BO_H   262144u      // bf16 [128][1024]  256 KiB
#define BO_Q   524288u      // bf16 [64][1024]   128 KiB (memory-slot queries only)
#define BO_K   655360u      // bf16 [128][512]   128 KiB (row-major == QK B-operand)
#define BO_VT  786432u      // bf16 [8][64][128] 128 KiB (transposed == PV B-operand)
#define BO_ENC 917504u      // bf16 [64][1024]   128 KiB
#define BO_OUT 1048576u     // f32  [64][1024]   256 KiB

// ---------------------------------------------------------------------------
// device helpers
// ---------------------------------------------------------------------------

__device__ __forceinline__ v8f zero8() {
    v8f z;
#pragma unroll
    for (int i = 0; i < 8; ++i) z[i] = 0.0f;
    return z;
}

__device__ __forceinline__ v8f wmma_bf16(v16bf a, v16bf b, v8f c) {
    // emits v_wmma_f32_16x16x32_bf16
    return __builtin_amdgcn_wmma_f32_16x16x32_bf16(false, a, false, b, (short)0, c,
                                                   false, false);
}

// A-operand fragment (16x32 bf16, contiguous k order used here; B packing matches):
//   lanes 0-15 : M=lane,    elems 0-15 -> K+0..15
//   lanes 16-31: M=lane-16, elems 0-15 -> K+16..31
// p points at (row, k0); koff = (lane<16)?0:8.
__device__ __forceinline__ v16bf load_a(const bf16_t* p, int koff) {
    v8bf lo = *(const v8bf*)(p + 2 * koff);
    v8bf hi = *(const v8bf*)(p + 2 * koff + 8);
    return __builtin_shufflevector(lo, hi, 0, 1, 2, 3, 4, 5, 6, 7,
                                   8, 9, 10, 11, 12, 13, 14, 15);
}

__device__ __forceinline__ float wave_sum32(float v) {
#pragma unroll
    for (int m = 16; m >= 1; m >>= 1) v += __shfl_xor(v, m, 32);
    return v;
}
__device__ __forceinline__ float half_max16(float v) {
#pragma unroll
    for (int m = 8; m >= 1; m >>= 1) v = fmaxf(v, __shfl_xor(v, m, 32));
    return v;
}
__device__ __forceinline__ float half_sum16(float v) {
#pragma unroll
    for (int m = 8; m >= 1; m >>= 1) v += __shfl_xor(v, m, 32);
    return v;
}

__device__ __forceinline__ void phase_sync() {
    __threadfence_block();
    __syncthreads();
}

// ---------------------------------------------------------------------------
// prep kernels
// ---------------------------------------------------------------------------

// Repack fp32 W[K][N] (row-major) into WMMA B-operand layout:
// dst[((kt*NT + nt)*32 + lane)*16 + e] = W[kt*32 + (lane<16?0:16) + e][nt*16 + (lane&15)]
__global__ void pack_w_kernel(const float* __restrict__ src, bf16_t* __restrict__ dst,
                              int K, int N) {
    int idx = blockIdx.x * blockDim.x + threadIdx.x;
    if (idx >= K * N) return;
    int e = idx & 15;
    int l = (idx >> 4) & 31;
    int rest = idx >> 9;
    int NT = N >> 4;
    int nt = rest % NT;
    int kt = rest / NT;
    int k = kt * 32 + ((l < 16) ? 0 : 16) + e;
    int n = nt * 16 + (l & 15);
    dst[idx] = (bf16_t)src[k * N + n];
}

__global__ void rope_kernel(float* __restrict__ rc, float* __restrict__ rs) {
    int idx = blockIdx.x * blockDim.x + threadIdx.x;
    if (idx >= S_ * 32) return;
    int pos = idx >> 5;
    int i = idx & 31;
    float freq = __powf(10000.0f, -(2.0f * (float)i) / (float)HD_);
    float a = (float)pos * freq;
    rc[idx] = __cosf(a);
    rs[idx] = __sinf(a);
}

__global__ void meminit_kernel(const float* __restrict__ mem_state, char* __restrict__ base) {
    int idx = blockIdx.x * blockDim.x + threadIdx.x;
    if (idx >= B_ * MEM_ * D_) return;
    int b = idx >> 16;          // 65536 = MEM_*D_
    int d = idx & 65535;
    ((float*)(base + (size_t)b * PER_BATCH + BO_MEM))[d] = mem_state[d];
}

// ---------------------------------------------------------------------------
// main persistent kernel: one block per batch, 16 waves, 64-step recurrence
// ---------------------------------------------------------------------------

__global__ __launch_bounds__(NTHREADS, 1) void rmt_main_kernel(
    const float* __restrict__ hidden,        // [B][NM][MB][D]
    const unsigned char* __restrict__ mask,  // [B][NM] (jnp bool_, 1 byte)
    const bf16_t* __restrict__ wq_p, const bf16_t* __restrict__ wk_p,
    const bf16_t* __restrict__ wv_p, const bf16_t* __restrict__ wo_p,
    const float* __restrict__ ropec, const float* __restrict__ ropes,
    const float* __restrict__ pre_scale, const float* __restrict__ pre_bias,
    const float* __restrict__ post_scale, const float* __restrict__ post_bias,
    char* __restrict__ batch_base, float* __restrict__ out) {

    __shared__ __align__(32) bf16_t att_lds[NWAVE * 16 * 128];  // 64 KiB

    const int b = blockIdx.x;
    const int wave = threadIdx.x >> 5;
    const int lane = threadIdx.x & 31;
    const int ln = lane & 15;
    const int rowoff = (lane < 16) ? 0 : 8;   // C-fragment row offset
    const int koffA = (lane < 16) ? 0 : 8;    // A-fragment K offset
    const int koffB = (lane < 16) ? 0 : 16;   // B-fragment K offset

    char* bb = batch_base + (size_t)b * PER_BATCH;
    float*  memb = (float*)(bb + BO_MEM);
    bf16_t* hb   = (bf16_t*)(bb + BO_H);
    bf16_t* qb   = (bf16_t*)(bb + BO_Q);
    bf16_t* kb2  = (bf16_t*)(bb + BO_K);
    bf16_t* vT   = (bf16_t*)(bb + BO_VT);
    bf16_t* encb = (bf16_t*)(bb + BO_ENC);
    float*  outb = (float*)(bb + BO_OUT);

    for (int step = 0; step < NM_; ++step) {
        const float* xstep = hidden + (((size_t)b * NM_ + step) * MB_) * D_;

        // ---- Phase 1: pre-LN of concat(x, mem) -> h (bf16) -------------
#pragma unroll 1
        for (int i = 0; i < 8; ++i) {
            int row = wave * 8 + i;
            const float* src = (row < MB_) ? (xstep + (size_t)row * D_)
                                           : (memb + (size_t)(row - MB_) * D_);
            float vals[32];
            float s = 0.0f, sq = 0.0f;
#pragma unroll
            for (int j = 0; j < 32; ++j) {
                float v = src[j * 32 + lane];
                vals[j] = v;
                s += v;
                sq += v * v;
            }
            s = wave_sum32(s);
            sq = wave_sum32(sq);
            float mean = s * (1.0f / D_);
            float var = sq * (1.0f / D_) - mean * mean;
            float rstd = rsqrtf(var + 1e-6f);
#pragma unroll
            for (int j = 0; j < 32; ++j) {
                int c = j * 32 + lane;
                hb[(size_t)row * D_ + c] =
                    (bf16_t)((vals[j] - mean) * rstd * pre_scale[c] + pre_bias[c]);
            }
        }
        phase_sync();

        // ---- Phase 2: QKV GEMMs (WMMA bf16, 4x N-blocked) + RoPE --------
        // quad-tiles (4 consecutive 16-col tiles): q 4x16=64, k 8x8=64, v 64
#pragma unroll 1
        for (int t = wave; t < 192; t += NWAVE) {
            int kind, tm, ntq, NT, hrow0;
            const bf16_t* wp;
            if (t < 64)       { kind = 0; tm = t >> 4;         ntq = t & 15;        NT = 64; wp = wq_p; hrow0 = MB_ + tm * 16; }
            else if (t < 128) { kind = 1; tm = (t - 64) >> 3;  ntq = (t - 64) & 7;  NT = 32; wp = wk_p; hrow0 = tm * 16; }
            else              { kind = 2; tm = (t - 128) >> 3; ntq = (t - 128) & 7; NT = 32; wp = wv_p; hrow0 = tm * 16; }
            const int nt0 = ntq * 4;

            v8f acc[4];
#pragma unroll
            for (int u = 0; u < 4; ++u) acc[u] = zero8();
            const bf16_t* arow = hb + (size_t)(hrow0 + ln) * D_;
#pragma unroll 4
            for (int kt = 0; kt < 32; ++kt) {
                v16bf a = load_a(arow + kt * 32, koffA);   // shared across 4 B tiles
                v16bf bfr[4];
#pragma unroll
                for (int u = 0; u < 4; ++u)
                    bfr[u] = *(const v16bf*)(wp + ((size_t)(kt * NT + nt0 + u) * 32 + lane) * 16);
#pragma unroll
                for (int u = 0; u < 4; ++u) acc[u] = wmma_bf16(a, bfr[u], acc[u]);
            }

#pragma unroll
            for (int u = 0; u < 4; ++u) {
                int col = (nt0 + u) * 16 + ln;
                if (kind == 2) {  // v -> transposed store [kv][d][s]
                    int kv = col >> 6, dd = col & 63;
#pragma unroll
                    for (int r = 0; r < 8; ++r) {
                        int srow = tm * 16 + rowoff + r;
                        vT[(size_t)kv * (HD_ * S_) + (size_t)dd * S_ + srow] = (bf16_t)acc[u][r];
                    }
                } else {          // q or k -> RoPE then row-major store
                    int dd = col & 63;
#pragma unroll
                    for (int r = 0; r < 8; ++r) {
                        float x = acc[u][r];
                        float other = __shfl_xor(x, 1, 32);  // (even,odd) RoPE partner
                        int row = tm * 16 + rowoff + r;
                        int pos = (kind == 0) ? (MB_ + row) : row;
                        float c = ropec[pos * 32 + (dd >> 1)];
                        float sn = ropes[pos * 32 + (dd >> 1)];
                        float res = (dd & 1) ? fmaf(other, sn, x * c)
                                             : fmaf(-other, sn, x * c);
                        if (kind == 0) qb[(size_t)row * D_ + col]    = (bf16_t)res;
                        else           kb2[(size_t)row * KVD_ + col] = (bf16_t)res;
                    }
                }
            }
        }
        phase_sync();

        // ---- Phase 3: attention, one head per wave ----------------------
        {
            const int head = wave;
            const int wkv = head >> 1;  // GQA: repeat(k, 2)
            bf16_t* attw = att_lds + head * (16 * 128);
            const bf16_t* vh = vT + (size_t)wkv * (HD_ * S_);
#pragma unroll 1
            for (int mt = 0; mt < 4; ++mt) {
                // logits: 16 q rows x 128 keys, K = HD = 64
                v8f lg[8];
#pragma unroll
                for (int nt = 0; nt < 8; ++nt) lg[nt] = zero8();
#pragma unroll
                for (int kbk = 0; kbk < 2; ++kbk) {
                    v16bf a = load_a(qb + (size_t)(mt * 16 + ln) * D_ + head * HD_ + kbk * 32,
                                     koffA);
#pragma unroll
                    for (int nt = 0; nt < 8; ++nt) {
                        v16bf bfr = *(const v16bf*)(kb2 + (size_t)(nt * 16 + ln) * KVD_ +
                                                    wkv * HD_ + kbk * 32 + koffB);
                        lg[nt] = wmma_bf16(a, bfr, lg[nt]);
                    }
                }
                // softmax (scale = 1/sqrt(64) = 0.125), rows live in 16-lane halves
#pragma unroll
                for (int r = 0; r < 8; ++r) {
                    float mx = -1e30f;
#pragma unroll
                    for (int nt = 0; nt < 8; ++nt) {
                        float v = lg[nt][r] * 0.125f;
                        lg[nt][r] = v;
                        mx = fmaxf(mx, v);
                    }
                    mx = half_max16(mx);
                    float sum = 0.0f;
#pragma unroll
                    for (int nt = 0; nt < 8; ++nt) {
                        float e = __expf(lg[nt][r] - mx);
                        lg[nt][r] = e;
                        sum += e;
                    }
                    sum = half_sum16(sum);
                    float inv = 1.0f / sum;
#pragma unroll
                    for (int nt = 0; nt < 8; ++nt)
                        attw[(rowoff + r) * 128 + nt * 16 + ln] = (bf16_t)(lg[nt][r] * inv);
                }
                // wave-private LDS RAW: CDNA5 split counter wait
                asm volatile("s_wait_dscnt 0x0" ::: "memory");

                // PV: 16 q rows x 64 d, K = 128 keys
                v8f ev[4];
#pragma unroll
                for (int nt = 0; nt < 4; ++nt) ev[nt] = zero8();
#pragma unroll
                for (int kbk = 0; kbk < 4; ++kbk) {
                    v16bf a = load_a(attw + ln * 128 + kbk * 32, koffA);
#pragma unroll
                    for (int nt = 0; nt < 4; ++nt) {
                        v16bf bfr = *(const v16bf*)(vh + (size_t)(nt * 16 + ln) * S_ +
                                                    kbk * 32 + koffB);
                        ev[nt] = wmma_bf16(a, bfr, ev[nt]);
                    }
                }
#pragma unroll
                for (int nt = 0; nt < 4; ++nt)
#pragma unroll
                    for (int r = 0; r < 8; ++r)
                        encb[(size_t)(mt * 16 + rowoff + r) * D_ + head * HD_ + nt * 16 + ln] =
                            (bf16_t)ev[nt][r];
            }
        }
        phase_sync();

        // ---- Phase 4: out = enc @ wo (4x N-blocked) ---------------------
#pragma unroll 1
        for (int t = wave; t < 64; t += NWAVE) {
            int tm = t >> 4, ntq = t & 15;
            const int nt0 = ntq * 4;
            v8f acc[4];
#pragma unroll
            for (int u = 0; u < 4; ++u) acc[u] = zero8();
            const bf16_t* arow = encb + (size_t)(tm * 16 + ln) * D_;
#pragma unroll 4
            for (int kt = 0; kt < 32; ++kt) {
                v16bf a = load_a(arow + kt * 32, koffA);
                v16bf bfr[4];
#pragma unroll
                for (int u = 0; u < 4; ++u)
                    bfr[u] = *(const v16bf*)(wo_p + ((size_t)(kt * 64 + nt0 + u) * 32 + lane) * 16);
#pragma unroll
                for (int u = 0; u < 4; ++u) acc[u] = wmma_bf16(a, bfr[u], acc[u]);
            }
#pragma unroll
            for (int u = 0; u < 4; ++u)
#pragma unroll
                for (int r = 0; r < 8; ++r)
                    outb[(size_t)(tm * 16 + rowoff + r) * D_ + (nt0 + u) * 16 + ln] = acc[u][r];
        }
        phase_sync();

        // ---- Phase 5: post-LN(out + mem), masked state update -----------
        {
            bool keep = (mask[(size_t)b * NM_ + step] != 0);
#pragma unroll 1
            for (int i = 0; i < 4; ++i) {
                int row = wave * 4 + i;
                float vals[32];
                float s = 0.0f, sq = 0.0f;
#pragma unroll
                for (int j = 0; j < 32; ++j) {
                    int c = j * 32 + lane;
                    float y = outb[(size_t)row * D_ + c] + memb[(size_t)row * D_ + c];
                    vals[j] = y;
                    s += y;
                    sq += y * y;
                }
                s = wave_sum32(s);
                sq = wave_sum32(sq);
                float mean = s * (1.0f / D_);
                float var = sq * (1.0f / D_) - mean * mean;
                float rstd = rsqrtf(var + 1e-6f);
                if (keep) {
#pragma unroll
                    for (int j = 0; j < 32; ++j) {
                        int c = j * 32 + lane;
                        memb[(size_t)row * D_ + c] =
                            (vals[j] - mean) * rstd * post_scale[c] + post_bias[c];
                    }
                }
            }
            // prefetch next step's hidden block (64x1024 f32 = 256 KiB);
            // 512 threads x 4 x 128B lines -> global_prefetch_b8
            if (step + 1 < NM_) {
                const char* xnext = (const char*)(xstep + (size_t)MB_ * D_);
#pragma unroll
                for (int p = 0; p < 4; ++p)
                    __builtin_prefetch(xnext + ((size_t)threadIdx.x * 4 + p) * 128, 0, 1);
            }
        }
        phase_sync();
    }

    // ---- final: write memory state to d_out ----------------------------
#pragma unroll 1
    for (int i = threadIdx.x; i < MEM_ * D_; i += NTHREADS)
        out[(size_t)b * (MEM_ * D_) + i] = memb[i];
}

// ---------------------------------------------------------------------------
// host launcher
// ---------------------------------------------------------------------------

extern "C" void kernel_launch(void* const* d_in, const int* in_sizes, int n_in,
                              void* d_out, int out_size, void* d_ws, size_t ws_size,
                              hipStream_t stream) {
    const float* hidden     = (const float*)d_in[0];
    const unsigned char* mk = (const unsigned char*)d_in[1];  // jnp bool_ = 1 byte
    const float* mem_state  = (const float*)d_in[2];
    const float* wq         = (const float*)d_in[3];
    const float* wk         = (const float*)d_in[4];
    const float* wv         = (const float*)d_in[5];
    const float* wo         = (const float*)d_in[6];
    const float* pre_scale  = (const float*)d_in[7];
    const float* pre_bias   = (const float*)d_in[8];
    const float* post_scale = (const float*)d_in[9];
    const float* post_bias  = (const float*)d_in[10];

    char* ws = (char*)d_ws;
    bf16_t* wq_p = (bf16_t*)(ws + OFF_WQ);
    bf16_t* wk_p = (bf16_t*)(ws + OFF_WK);
    bf16_t* wv_p = (bf16_t*)(ws + OFF_WV);
    bf16_t* wo_p = (bf16_t*)(ws + OFF_WO);
    float* ropec = (float*)(ws + OFF_RC);
    float* ropes = (float*)(ws + OFF_RS);
    char* batch_base = ws + OFF_BATCH;

    // prep: weight repack to WMMA B layout, rope tables, per-batch mem init
    pack_w_kernel<<<(D_ * D_ + 255) / 256, 256, 0, stream>>>(wq, wq_p, D_, D_);
    pack_w_kernel<<<(D_ * KVD_ + 255) / 256, 256, 0, stream>>>(wk, wk_p, D_, KVD_);
    pack_w_kernel<<<(D_ * KVD_ + 255) / 256, 256, 0, stream>>>(wv, wv_p, D_, KVD_);
    pack_w_kernel<<<(D_ * D_ + 255) / 256, 256, 0, stream>>>(wo, wo_p, D_, D_);
    rope_kernel<<<(S_ * 32 + 255) / 256, 256, 0, stream>>>(ropec, ropes);
    meminit_kernel<<<(B_ * MEM_ * D_ + 255) / 256, 256, 0, stream>>>(mem_state, batch_base);

    // persistent recurrence: one block per batch
    rmt_main_kernel<<<B_, NTHREADS, 0, stream>>>(
        hidden, mk, wq_p, wk_p, wv_p, wo_p, ropec, ropes,
        pre_scale, pre_bias, post_scale, post_bias, batch_base, (float*)d_out);
}
